// LocalSelfAttention_53790170415632
// MI455X (gfx1250) — matrix operand order repaired and checked by reading, not compiled, not run on hardware
//
#include <hip/hip_runtime.h>
#include <math.h>

// ---------------------------------------------------------------------------
// Types for CDNA5 WMMA (wave32, 16x16x32 bf16 -> f32)
// ---------------------------------------------------------------------------
typedef __bf16 bf16;
typedef bf16 v16bf __attribute__((ext_vector_type(16)));
typedef float v8f __attribute__((ext_vector_type(8)));
typedef unsigned int u32x2 __attribute__((ext_vector_type(2)));
typedef unsigned int u32x4 __attribute__((ext_vector_type(4)));

static __device__ __forceinline__ v8f wmma_bf16(v16bf a, v16bf b, v8f c) {
  return __builtin_amdgcn_wmma_f32_16x16x32_bf16(false, a, false, b, (short)0,
                                                 c, false, false);
}

// pack two f32 -> two bf16 in one dword (v_cvt_pk_bf16_f32)
static __device__ __forceinline__ unsigned pk2(float a, float b) {
  bf16 x = (bf16)a, y = (bf16)b;
  unsigned short ux = __builtin_bit_cast(unsigned short, x);
  unsigned short uy = __builtin_bit_cast(unsigned short, y);
  return (unsigned)ux | ((unsigned)uy << 16);
}

// ---------------------------------------------------------------------------
// Generic tiled GEMM: C[M,N] = A[M,K] @ B[K,N] + bias, bf16 WMMA, f32 accum.
//   mode 0: scatter columns into Q/K/V buffers laid out [B,H,S,D]
//   mode 1: C = acc + bias + resid (row stride N), write Cout
// Block: 128 threads (4 waves). Tile 128x128, K-step 32; each wave computes a
// 32x128 strip (2 A-subtiles x 8 B-tiles = 16 WMMA/step).
// Software pipeline: global loads for step ks+32 are issued into staging regs
// before the WMMA chain of step ks; only cvt+ds_store sit between barriers.
// LDS holds A/B tiles pre-swizzled into WMMA operand layouts:
//   A (16x32): lane = g*16+m holds 16 contiguous bf16 (K = g*8+[0..7], 16+g*8+[0..7])
//   B (32x16): lane = g*16+n holds 16 contiguous bf16 (K = g*8+[0..7], 16+g*8+[0..7])
// ---------------------------------------------------------------------------
__global__ __launch_bounds__(128) void gemm_bf16_kernel(
    const float* __restrict__ A, const float* __restrict__ Bm,
    const float* __restrict__ bias, const float* __restrict__ resid,
    float* __restrict__ Cout, float* __restrict__ Qb, float* __restrict__ Kb,
    float* __restrict__ Vb, int M, int N, int K, int mode) {
  __shared__ __align__(32) unsigned short sA[8 * 512];  // 8 KB
  __shared__ __align__(32) unsigned short sB[8 * 512];  // 8 KB

  const int tid = threadIdx.x;
  const int wave = tid >> 5;
  const int lane = tid & 31;
  const int m0 = blockIdx.y * 128;
  const int n0 = blockIdx.x * 128;

  float4 stA[8];
  float stB[8][4];

  // ---- staging-register loaders (global -> regs) ----
  auto loadA = [&](int ks) {
#pragma unroll
    for (int i = 0; i < 8; ++i) {
      int gidx = tid + i * 128;  // 0..1023
      int row = gidx >> 3;       // 0..127
      int kk = (gidx & 7) * 4;   // 0..28
      const float* ap = A + (size_t)(m0 + row) * K + ks + kk;
      stA[i] = *(const float4*)ap;
      if (ks + 64 < K) __builtin_prefetch(ap + 64, 0, 1);
    }
  };
  auto loadB = [&](int ks) {
#pragma unroll
    for (int i = 0; i < 8; ++i) {
      int gidx = tid + i * 128;   // 0..1023
      int col = gidx & 127;
      int kk0 = (gidx >> 7) * 4;  // 0..28
      const float* bp = Bm + (size_t)(ks + kk0) * N + n0 + col;
      stB[i][0] = bp[0];
      stB[i][1] = bp[(size_t)N];
      stB[i][2] = bp[2 * (size_t)N];
      stB[i][3] = bp[3 * (size_t)N];
      if (ks + 64 < K) __builtin_prefetch(bp + 64 * (size_t)N, 0, 1);
    }
  };
  // ---- staging regs -> LDS (bf16, operand-swizzled) ----
  auto storeAB = [&]() {
#pragma unroll
    for (int i = 0; i < 8; ++i) {
      int gidx = tid + i * 128;
      int row = gidx >> 3;
      int kk = (gidx & 7) * 4;
      int g = (kk & 15) >> 3;
      int e0 = ((kk >> 4) << 3) + (kk & 7);
      int lanei = g * 16 + (row & 15);
      int idx = ((row >> 4) << 9) + (lanei << 4) + e0;
      *(u32x2*)(&sA[idx]) =
          (u32x2){pk2(stA[i].x, stA[i].y), pk2(stA[i].z, stA[i].w)};
    }
#pragma unroll
    for (int i = 0; i < 8; ++i) {
      int gidx = tid + i * 128;
      int col = gidx & 127;
      int kk0 = (gidx >> 7) * 4;
      int g = (kk0 & 15) >> 3;
      int e0 = ((kk0 >> 4) << 3) + (kk0 & 7);
      int lanei = g * 16 + (col & 15);
      int idx = ((col >> 4) << 9) + (lanei << 4) + e0;
      *(u32x2*)(&sB[idx]) =
          (u32x2){pk2(stB[i][0], stB[i][1]), pk2(stB[i][2], stB[i][3])};
    }
  };

  v8f acc0[8], acc1[8];
#pragma unroll
  for (int i = 0; i < 8; ++i) {
    acc0[i] = (v8f){};
    acc1[i] = (v8f){};
  }

  loadA(0);
  loadB(0);
  for (int ks = 0; ks < K; ks += 32) {
    __syncthreads();  // previous step done reading LDS
    storeAB();
    __syncthreads();
    if (ks + 32 < K) {  // issue next step's global loads under the WMMAs
      loadA(ks + 32);
      loadB(ks + 32);
    }
    // ---- compute: wave = 32-row strip (subtiles 2w, 2w+1) x 128 cols ----
    v16bf a0 = *(const v16bf*)(&sA[((2 * wave) << 9) + (lane << 4)]);
    v16bf a1 = *(const v16bf*)(&sA[((2 * wave + 1) << 9) + (lane << 4)]);
    v16bf bcur = *(const v16bf*)(&sB[lane << 4]);
#pragma unroll
    for (int ct = 0; ct < 8; ++ct) {
      v16bf bnext =
          *(const v16bf*)(&sB[(((ct + 1) & 7) << 9) + (lane << 4)]);
      acc0[ct] = wmma_bf16(a0, bcur, acc0[ct]);
      acc1[ct] = wmma_bf16(a1, bcur, acc1[ct]);
      bcur = bnext;
    }
  }

  // ---- epilogue ----
  const int grp = lane >> 4;
  const int nn = lane & 15;
#pragma unroll
  for (int sub = 0; sub < 2; ++sub) {
    v8f* accp = sub ? acc1 : acc0;
#pragma unroll
    for (int ct = 0; ct < 8; ++ct) {
      int ncol = n0 + ct * 16 + nn;
      float bv = bias ? bias[ncol] : 0.0f;
#pragma unroll
      for (int r = 0; r < 8; ++r) {
        int mr = m0 + wave * 32 + sub * 16 + grp * 8 + r;
        float val = accp[ct][r] + bv;
        if (mode == 0) {
          // scatter qkv column -> [B,H,S,D] buffers
          int bidx = mr >> 11;     // S = 2048
          int s = mr & 2047;
          int which = ncol >> 10;  // DM = 1024
          int cw = ncol & 1023;
          int h = cw >> 6;         // D = 64
          int d = cw & 63;
          float* dst = (which == 0) ? Qb : (which == 1) ? Kb : Vb;
          dst[(((size_t)bidx * 16 + h) * 2048 + s) * 64 + d] = val;
        } else {
          val += resid[(size_t)mr * N + ncol];
          Cout[(size_t)mr * N + ncol] = val;
        }
      }
    }
  }
}

// ---------------------------------------------------------------------------
// RoPE (interleaved pairs), in place on Q and K [B,H,S,D] buffers.
// one thread per (which, bh, s, pair)
// ---------------------------------------------------------------------------
__global__ __launch_bounds__(256) void rope_kernel(float* __restrict__ Q,
                                                   float* __restrict__ K) {
  int tid = blockIdx.x * 256 + threadIdx.x;  // 2^22 total
  int pair = tid & 31;
  int s = (tid >> 5) & 2047;
  int bh = (tid >> 16) & 31;
  int which = tid >> 21;
  float* base = (which ? K : Q) + ((size_t)bh * 2048 + s) * 64 + pair * 2;
  float inv_freq = powf(10000.0f, -(float)(2 * pair) / 64.0f);
  float ang = (float)s * inv_freq;
  float sn, cs;
  sincosf(ang, &sn, &cs);
  float x0 = base[0], x1 = base[1];
  base[0] = x0 * cs - x1 * sn;
  base[1] = x0 * sn + x1 * cs;
}

// ---------------------------------------------------------------------------
// Flash attention, bf16 WMMA, online softmax.
// Block: 128 threads (4 waves). Each block: one (b,h), 64 query rows.
// Each wave owns 16 q-rows; Q kept resident in registers as A-operands.
// Per 32-key tile: 4 WMMA (scores, chained over D=64) + 4 WMMA (P@V).
// K/V global loads for tile kt+32 are issued before the compute of tile kt.
// grid: (S/64, B*H)
// ---------------------------------------------------------------------------
__global__ __launch_bounds__(128) void attn_kernel(const float* __restrict__ Qg,
                                                   const float* __restrict__ Kg,
                                                   const float* __restrict__ Vg,
                                                   float* __restrict__ Og) {
  const int S = 2048, D = 64;
  const int bh = blockIdx.y;
  const int qb = blockIdx.x;
  const int tid = threadIdx.x;
  const int wave = tid >> 5;
  const int lane = tid & 31;
  const int grp = lane >> 4;
  const int nn = lane & 15;

  const float* Qbase = Qg + (size_t)bh * S * D;
  const float* Kbase = Kg + (size_t)bh * S * D;
  const float* Vbase = Vg + (size_t)bh * S * D;

  __shared__ __align__(32) unsigned short sQ[4 * 1024];  // per-wave A-operands
  __shared__ __align__(32) unsigned short sK[4 * 512];   // (dchunk,keySub) B-ops
  __shared__ __align__(32) unsigned short sV[4 * 512];   // per d-subtile B-ops
  __shared__ __align__(32) unsigned short sP[4 * 512];   // per-wave P 16x32

  // ---- load Q once (scaled by 1/sqrt(D)=0.125), A-operand layout ----
  const int q0 = qb * 64;
#pragma unroll
  for (int i = 0; i < 8; ++i) {
    int gidx = tid + i * 128;  // 0..1023
    int row = gidx >> 4;       // 0..63
    int dg = (gidx & 15) * 4;  // 0..60
    float4 v = *(const float4*)(Qbase + (size_t)(q0 + row) * D + dg);
    int wch = dg >> 5;
    int dd = dg & 31;
    int g = (dd & 15) >> 3;
    int e0 = ((dd >> 4) << 3) + (dd & 7);
    int lanei = g * 16 + (row & 15);
    int idx = (row >> 4) * 1024 + wch * 512 + lanei * 16 + e0;
    *(u32x2*)(&sQ[idx]) = (u32x2){pk2(v.x * 0.125f, v.y * 0.125f),
                                  pk2(v.z * 0.125f, v.w * 0.125f)};
  }
  __syncthreads();
  v16bf a0 = *(const v16bf*)(&sQ[wave * 1024 + lane * 16]);
  v16bf a1 = *(const v16bf*)(&sQ[wave * 1024 + 512 + lane * 16]);

  v8f o0 = (v8f){}, o1 = (v8f){}, o2 = (v8f){}, o3 = (v8f){};
  float Mrow[8], Lrow[8];
#pragma unroll
  for (int r = 0; r < 8; ++r) {
    Mrow[r] = -1e30f;
    Lrow[r] = 0.0f;
  }

  // ---- K/V tile staging (global -> regs -> swizzled LDS) ----
  float4 stK[4];
  float stV[4][4];
  auto loadKV = [&](int kt) {
#pragma unroll
    for (int i = 0; i < 4; ++i) {
      int gidx = tid + i * 128;  // 0..511
      int key = gidx >> 4;       // 0..31
      int dg = (gidx & 15) * 4;  // 0..60
      stK[i] = *(const float4*)(Kbase + (size_t)(kt + key) * D + dg);
    }
#pragma unroll
    for (int i = 0; i < 4; ++i) {
      int gidx = tid + i * 128;  // 0..511
      int d = gidx & 63;
      int k4 = (gidx >> 6) * 4;  // 0..28
      const float* vp = Vbase + (size_t)(kt + k4) * D + d;
      stV[i][0] = vp[0];
      stV[i][1] = vp[D];
      stV[i][2] = vp[2 * D];
      stV[i][3] = vp[3 * D];
    }
  };
  auto storeKV = [&]() {
#pragma unroll
    for (int i = 0; i < 4; ++i) {
      int gidx = tid + i * 128;
      int key = gidx >> 4;
      int dg = (gidx & 15) * 4;
      int wch = dg >> 5;
      int dd = dg & 31;
      int g = (dd & 15) >> 3;
      int e0 = ((dd >> 4) << 3) + (dd & 7);
      int lanei = g * 16 + (key & 15);
      int idx = (wch * 2 + (key >> 4)) * 512 + lanei * 16 + e0;
      *(u32x2*)(&sK[idx]) =
          (u32x2){pk2(stK[i].x, stK[i].y), pk2(stK[i].z, stK[i].w)};
    }
#pragma unroll
    for (int i = 0; i < 4; ++i) {
      int gidx = tid + i * 128;
      int d = gidx & 63;
      int k4 = (gidx >> 6) * 4;
      int g = (k4 & 15) >> 3;
      int e0 = ((k4 >> 4) << 3) + (k4 & 7);
      int lanei = g * 16 + (d & 15);
      int idx = (d >> 4) * 512 + lanei * 16 + e0;
      *(u32x2*)(&sV[idx]) =
          (u32x2){pk2(stV[i][0], stV[i][1]), pk2(stV[i][2], stV[i][3])};
    }
  };

  loadKV(0);
  for (int kt = 0; kt < S; kt += 32) {
    __syncthreads();  // previous tile done reading sK/sV
    storeKV();
    __syncthreads();
    if (kt + 32 < S) loadKV(kt + 32);  // overlap with compute below

    // ---- scores: S16x32 = Q(16x64) @ K^T(64x32), 4 WMMA ----
    v8f c0 = (v8f){}, c1 = (v8f){};
    {
      v16bf b00 = *(const v16bf*)(&sK[0 * 512 + lane * 16]);
      v16bf b01 = *(const v16bf*)(&sK[1 * 512 + lane * 16]);
      v16bf b10 = *(const v16bf*)(&sK[2 * 512 + lane * 16]);
      v16bf b11 = *(const v16bf*)(&sK[3 * 512 + lane * 16]);
      c0 = wmma_bf16(a0, b00, c0);
      c0 = wmma_bf16(a1, b10, c0);
      c1 = wmma_bf16(a0, b01, c1);
      c1 = wmma_bf16(a1, b11, c1);
    }

    // ---- online softmax (16-lane butterfly on C layout) ----
    float alpha[8];
#pragma unroll
    for (int r = 0; r < 8; ++r) {
      float mx = fmaxf(c0[r], c1[r]);
      mx = fmaxf(mx, __shfl_xor(mx, 1));
      mx = fmaxf(mx, __shfl_xor(mx, 2));
      mx = fmaxf(mx, __shfl_xor(mx, 4));
      mx = fmaxf(mx, __shfl_xor(mx, 8));
      float mnew = fmaxf(Mrow[r], mx);
      alpha[r] = __expf(Mrow[r] - mnew);
      Mrow[r] = mnew;
      float p0 = __expf(c0[r] - mnew);
      float p1 = __expf(c1[r] - mnew);
      c0[r] = p0;
      c1[r] = p1;
      float sm = p0 + p1;
      sm += __shfl_xor(sm, 1);
      sm += __shfl_xor(sm, 2);
      sm += __shfl_xor(sm, 4);
      sm += __shfl_xor(sm, 8);
      Lrow[r] = Lrow[r] * alpha[r] + sm;
      o0[r] *= alpha[r];
      o1[r] *= alpha[r];
      o2[r] *= alpha[r];
      o3[r] *= alpha[r];
    }

    // ---- transpose P (C layout -> A operand) through per-wave LDS ----
#pragma unroll
    for (int r = 0; r < 8; ++r) {
      int m = grp * 8 + r;
      sP[wave * 512 + m * 32 + nn] =
          __builtin_bit_cast(unsigned short, (bf16)c0[r]);
      sP[wave * 512 + m * 32 + 16 + nn] =
          __builtin_bit_cast(unsigned short, (bf16)c1[r]);
    }
    union {
      v16bf v;
      u32x4 q[2];
    } pu;
    {
      int m = lane & 15;
      int g = lane >> 4;
      pu.q[0] = *(const u32x4*)(&sP[wave * 512 + m * 32 + g * 8]);
      pu.q[1] = *(const u32x4*)(&sP[wave * 512 + m * 32 + 16 + g * 8]);
    }

    // ---- O += P(16x32) @ V(32x64), 4 WMMA ----
    {
      v16bf bv0 = *(const v16bf*)(&sV[0 * 512 + lane * 16]);
      v16bf bv1 = *(const v16bf*)(&sV[1 * 512 + lane * 16]);
      v16bf bv2 = *(const v16bf*)(&sV[2 * 512 + lane * 16]);
      v16bf bv3 = *(const v16bf*)(&sV[3 * 512 + lane * 16]);
      o0 = wmma_bf16(pu.v, bv0, o0);
      o1 = wmma_bf16(pu.v, bv1, o1);
      o2 = wmma_bf16(pu.v, bv2, o2);
      o3 = wmma_bf16(pu.v, bv3, o3);
    }
  }

  // ---- epilogue: O /= L, store into attn buffer [B,S,H,D] ----
  const int b = bh >> 4;
  const int h = bh & 15;
#pragma unroll
  for (int r = 0; r < 8; ++r) {
    float inv = 1.0f / Lrow[r];
    int srow = q0 + wave * 16 + grp * 8 + r;
    float* orow = Og + (((size_t)b * 2048 + srow) * 16 + h) * 64;
    orow[0 * 16 + nn] = o0[r] * inv;
    orow[1 * 16 + nn] = o1[r] * inv;
    orow[2 * 16 + nn] = o2[r] * inv;
    orow[3 * 16 + nn] = o3[r] * inv;
  }
}

// ---------------------------------------------------------------------------
// Row LayerNorm over DM=1024, one block (256 threads) per row, in-place OK.
// ---------------------------------------------------------------------------
__global__ __launch_bounds__(256) void ln_kernel(float* __restrict__ H,
                                                 const float* __restrict__ gamma,
                                                 const float* __restrict__ beta,
                                                 float* __restrict__ out) {
  __shared__ float rs[8], rss[8];
  const int row = blockIdx.x;
  const float* hp = H + (size_t)row * 1024;
  float v[4];
  float s = 0.0f, ss = 0.0f;
#pragma unroll
  for (int i = 0; i < 4; ++i) {
    v[i] = hp[threadIdx.x + i * 256];
    s += v[i];
    ss += v[i] * v[i];
  }
  for (int m = 1; m < 32; m <<= 1) {
    s += __shfl_xor(s, m);
    ss += __shfl_xor(ss, m);
  }
  const int wave = threadIdx.x >> 5, lane = threadIdx.x & 31;
  if (lane == 0) {
    rs[wave] = s;
    rss[wave] = ss;
  }
  __syncthreads();
  if (wave == 0) {
    float a = (lane < 8) ? rs[lane] : 0.0f;
    float b = (lane < 8) ? rss[lane] : 0.0f;
    for (int m = 1; m < 8; m <<= 1) {
      a += __shfl_xor(a, m);
      b += __shfl_xor(b, m);
    }
    if (lane == 0) {
      rs[0] = a;
      rss[0] = b;
    }
  }
  __syncthreads();
  float mean = rs[0] * (1.0f / 1024.0f);
  float var = rss[0] * (1.0f / 1024.0f) - mean * mean;
  float rstd = rsqrtf(var + 1e-5f);
#pragma unroll
  for (int i = 0; i < 4; ++i) {
    int c = threadIdx.x + i * 256;
    out[(size_t)row * 1024 + c] = (v[i] - mean) * rstd * gamma[c] + beta[c];
  }
}

// ---------------------------------------------------------------------------
// Orchestration
// inputs: x, Wqkv, bqkv, Wo, bo, gamma, beta  (all f32)
// ---------------------------------------------------------------------------
extern "C" void kernel_launch(void* const* d_in, const int* in_sizes, int n_in,
                              void* d_out, int out_size, void* d_ws,
                              size_t ws_size, hipStream_t stream) {
  (void)in_sizes;
  (void)n_in;
  (void)out_size;
  (void)ws_size;
  const float* x = (const float*)d_in[0];
  const float* Wqkv = (const float*)d_in[1];
  const float* bqkv = (const float*)d_in[2];
  const float* Wo = (const float*)d_in[3];
  const float* bo = (const float*)d_in[4];
  const float* gamma = (const float*)d_in[5];
  const float* beta = (const float*)d_in[6];
  float* out = (float*)d_out;

  const size_t QKV_ELEMS = (size_t)2 * 16 * 2048 * 64;  // 4 Mi floats each
  float* q = (float*)d_ws;
  float* k = q + QKV_ELEMS;
  float* v = k + QKV_ELEMS;
  float* attn = v + QKV_ELEMS;  // [B,S,H,D] = [4096,1024]

  // 1) fused QKV projection (M=4096, N=3072, K=1024), scatter to q/k/v
  dim3 g1(3072 / 128, 4096 / 128);
  gemm_bf16_kernel<<<g1, 128, 0, stream>>>(x, Wqkv, bqkv, nullptr, nullptr, q,
                                           k, v, 4096, 3072, 1024, 0);
  // 2) RoPE on q, k
  rope_kernel<<<16384, 256, 0, stream>>>(q, k);
  // 3) flash attention -> attn [B,S,H*D]
  dim3 ga(2048 / 64, 32);
  attn_kernel<<<ga, 128, 0, stream>>>(q, k, v, attn);
  // 4) output projection + bias + residual -> h (in d_out)
  dim3 g3(1024 / 128, 4096 / 128);
  gemm_bf16_kernel<<<g3, 128, 0, stream>>>(attn, Wo, bo, x, out, nullptr,
                                           nullptr, nullptr, 4096, 1024, 1024,
                                           1);
  // 5) LayerNorm in place
  ln_kernel<<<4096, 256, 0, stream>>>(out, gamma, beta, out);
}
